// GroupQueryAttention_QKV_21139829030929
// MI455X (gfx1250) — matrix operand order repaired and checked by reading, not compiled, not run on hardware
//
#include <hip/hip_runtime.h>
#include <hip/hip_bf16.h>

// ---------------------------------------------------------------------------
// Fused RMSNorm + QKV projection (GQA) for MI455X (gfx1250, wave32, WMMA).
//
// M = B*S = 8192, K = H = 4096, N = 4096 (q) + 1024 (k) + 1024 (v) = 6144.
// ~412 GFLOP on ~420 MB unique bytes -> compute bound; bf16 WMMA, f32 acc.
// bf16 staging (xn 64MB + wt 48MB = 112MB) fits in the 192MB L2; GEMM output
// uses nontemporal stores so the 192MB result stream doesn't evict operands.
// GEMM stages tiles through LDS with CDNA5 async global->LDS copies
// (ASYNCcnt) double-buffered against WMMA compute.
// ---------------------------------------------------------------------------

typedef __attribute__((ext_vector_type(16))) __bf16 bf16x16;
typedef __attribute__((ext_vector_type(8)))  __bf16 bf16x8;
typedef __attribute__((ext_vector_type(4)))  __bf16 bf16x4;
typedef __attribute__((ext_vector_type(8)))  float  f32x8;

#define M_ROWS 8192
#define K_DIM  4096
#define N_Q    4096
#define N_KV   1024
#define N_TOT  6144

// GEMM tiling
#define BK        32
#define LDS_ROW   80                       // 64B data + 16B pad (16B aligned,
                                           // stride 20 dwords -> conflict-free)
#define TILE_B    (128 * LDS_ROW)          // 10240 B per tile (A or B)
#define BUF_B     (2 * TILE_B)             // A+B per pipeline stage

// ---------------------------------------------------------------------------
// Kernel 1: RMSNorm + fp32 -> bf16.  One block per row, 256 threads,
// 16 elements per thread (float4-vectorized), wave32 shuffle reduction.
// ---------------------------------------------------------------------------
__global__ __launch_bounds__(256) void rmsnorm_bf16_kernel(
    const float* __restrict__ x, const float* __restrict__ gamma,
    __bf16* __restrict__ xn) {
  const int row = blockIdx.x;
  const int tid = threadIdx.x;
  const float* xr = x + (size_t)row * K_DIM;

  float4 v[4];
  float ss = 0.0f;
#pragma unroll
  for (int i = 0; i < 4; ++i) {
    v[i] = reinterpret_cast<const float4*>(xr)[i * 256 + tid];
    ss += v[i].x * v[i].x + v[i].y * v[i].y + v[i].z * v[i].z + v[i].w * v[i].w;
  }
#pragma unroll
  for (int off = 16; off > 0; off >>= 1) ss += __shfl_xor(ss, off, 32);

  __shared__ float red[8];
  if ((tid & 31) == 0) red[tid >> 5] = ss;
  __syncthreads();
  float tot = 0.0f;
#pragma unroll
  for (int w = 0; w < 8; ++w) tot += red[w];

  const float rstd = rsqrtf(tot * (1.0f / (float)K_DIM) + 1e-6f);

  __bf16* xo = xn + (size_t)row * K_DIM;
#pragma unroll
  for (int i = 0; i < 4; ++i) {
    const int idx = i * 256 + tid;
    const float4 g = reinterpret_cast<const float4*>(gamma)[idx];
    bf16x4 o;
    o.x = (__bf16)(v[i].x * rstd * g.x);
    o.y = (__bf16)(v[i].y * rstd * g.y);
    o.z = (__bf16)(v[i].z * rstd * g.z);
    o.w = (__bf16)(v[i].w * rstd * g.w);
    reinterpret_cast<bf16x4*>(xo)[idx] = o;
  }
}

// ---------------------------------------------------------------------------
// Kernel 2: weight prep. Read w_q/w_k/w_v (fp32, [K, Nseg] row-major) and
// write a single bf16 K-major buffer wt[N_TOT][K_DIM] (wt[n][k] = W[k][n])
// via a 32x33 LDS transpose tile so both sides stay coalesced.
// ---------------------------------------------------------------------------
__global__ __launch_bounds__(256) void wprep_kernel(
    const float* __restrict__ wq, const float* __restrict__ wk,
    const float* __restrict__ wv, __bf16* __restrict__ wt) {
  __shared__ float tile[32][33];
  const int nb = blockIdx.x * 32;
  const int kb = blockIdx.y * 32;
  const int c  = threadIdx.x & 31;
  const int r0 = threadIdx.x >> 5;

  const float* src;
  int ldn, nloc;
  if (nb < N_Q) {
    src = wq; ldn = N_Q; nloc = nb;
  } else if (nb < N_Q + N_KV) {
    src = wk; ldn = N_KV; nloc = nb - N_Q;
  } else {
    src = wv; ldn = N_KV; nloc = nb - (N_Q + N_KV);
  }

#pragma unroll
  for (int r = r0; r < 32; r += 8)
    tile[r][c] = src[(size_t)(kb + r) * ldn + nloc + c];
  __syncthreads();
#pragma unroll
  for (int r = r0; r < 32; r += 8)
    wt[(size_t)(nb + r) * K_DIM + kb + c] = (__bf16)tile[c][r];
}

// ---------------------------------------------------------------------------
// Kernel 3: bf16 WMMA GEMM with async global->LDS double buffering.
// 256 threads = 8 waves; block tile 128x128, wave tile 32x64 (2x4 WMMAs).
// Pipeline per BK=32 step:
//   s_wait_asynccnt 0  -> this wave's DMA into cur buffer has landed
//   barrier            -> everyone's DMA landed; prev-buffer reads consumed
//   issue async copies for next buffer (overlaps with compute below)
//   ds_load fragments from cur buffer, 8x v_wmma_f32_16x16x32_bf16
// K loop is explicitly NOT unrolled (unroll 1) and the double-buffer select
// is a loop-carried XOR so accumulators stay in fixed registers: avoids the
// v_mov_b64 accumulator copies + WMMA->VALU hazard v_nops the 2x-unrolled
// version suffered.
// ---------------------------------------------------------------------------
__global__ __launch_bounds__(256) void qkv_gemm_kernel(
    const __bf16* __restrict__ xn, const __bf16* __restrict__ wt,
    float* __restrict__ out) {
  __shared__ __align__(16) char smem[2 * BUF_B];   // 40 KB

  const int tid   = threadIdx.x;
  const int lane  = tid & 31;
  const int wid   = tid >> 5;
  const int waveM = wid & 3;             // 4 waves down M
  const int waveN = wid >> 2;            // 2 waves across N
  const int mBase = blockIdx.y * 128;
  const int nBase = blockIdx.x * 128;
  const int l15   = lane & 15;
  const int hi    = lane >> 4;

  const unsigned ldsBase = (unsigned)(uintptr_t)(void*)smem;

  f32x8 acc[2][4] = {};

  // Cooperative loader: per stage, A tile (128x32 bf16 = 8KB) and B tile in
  // 16B chunks; 256 threads x 2 chunks x (A,B) = 4 async b128 per thread.
  auto issue_stage = [&](unsigned bufOff, int kb) {
#pragma unroll
    for (int j = 0; j < 2; ++j) {
      const int linear = tid + j * 256;        // 0..511
      const int row = linear >> 2;             // 0..127
      const int cid = linear & 3;              // 16B chunk within 64B row
      const unsigned ldsA = ldsBase + bufOff + row * LDS_ROW + cid * 16;
      const __bf16* ga = xn + (size_t)(mBase + row) * K_DIM + kb + cid * 8;
      asm volatile("global_load_async_to_lds_b128 %0, %1, off"
                   :: "v"(ldsA), "v"(ga) : "memory");
      const unsigned ldsB = ldsA + TILE_B;
      const __bf16* gb = wt + (size_t)(nBase + row) * K_DIM + kb + cid * 8;
      asm volatile("global_load_async_to_lds_b128 %0, %1, off"
                   :: "v"(ldsB), "v"(gb) : "memory");
    }
  };

  union Frag { bf16x16 v; bf16x8 h[2]; };

  issue_stage(0, 0);

  // Per-lane fragment offsets within a tile (constant over the loop).
  const unsigned aOff0 = (waveM * 32 + l15) * LDS_ROW + hi * 32;
  const unsigned bOff0 = TILE_B + (waveN * 64 + l15) * LDS_ROW + hi * 32;

  const int KSTEPS = K_DIM / BK;               // 128
  unsigned curOff = 0;                         // loop-carried buffer offset
#pragma unroll 1
  for (int it = 0; it < KSTEPS; ++it) {
    asm volatile("s_wait_asynccnt 0x0" ::: "memory");
    __syncthreads();
    const unsigned nxtOff = curOff ^ BUF_B;
    if (it + 1 < KSTEPS) issue_stage(nxtOff, (it + 1) * BK);

    const char* sbuf = smem + curOff;

    Frag a[2], b[4];
#pragma unroll
    for (int mt = 0; mt < 2; ++mt) {
      const char* p = sbuf + aOff0 + mt * (16 * LDS_ROW);
      a[mt].h[0] = *reinterpret_cast<const bf16x8*>(p);        // K hi*16 .. +7
      a[mt].h[1] = *reinterpret_cast<const bf16x8*>(p + 16);   // K hi*16+8 ..
    }
#pragma unroll
    for (int nt = 0; nt < 4; ++nt) {
      const char* p = sbuf + bOff0 + nt * (16 * LDS_ROW);
      b[nt].h[0] = *reinterpret_cast<const bf16x8*>(p);        // K hi*16 .. +7
      b[nt].h[1] = *reinterpret_cast<const bf16x8*>(p + 16);   // K hi*16+8 ..
    }
#pragma unroll
    for (int mt = 0; mt < 2; ++mt)
#pragma unroll
      for (int nt = 0; nt < 4; ++nt)
        acc[mt][nt] = __builtin_amdgcn_wmma_f32_16x16x32_bf16(
            false, a[mt].v, false, b[nt].v,
            (short)0, acc[mt][nt], false, false);

    curOff = nxtOff;
  }

  // Epilogue: scatter into q | k | v segments with nontemporal stores
  // (output stream is 192MB; keep the bf16 operands resident in L2).
  const size_t QSZ = (size_t)M_ROWS * N_Q;
  const size_t KSZ = (size_t)M_ROWS * N_KV;
  float* base;
  int ldo, nloc;
  const int nOff = nBase + waveN * 64;
  if (nOff < N_Q) {
    base = out; ldo = N_Q; nloc = nOff;
  } else if (nOff < N_Q + N_KV) {
    base = out + QSZ; ldo = N_KV; nloc = nOff - N_Q;
  } else {
    base = out + QSZ + KSZ; ldo = N_KV; nloc = nOff - (N_Q + N_KV);
  }

  const int mOff = mBase + waveM * 32;
#pragma unroll
  for (int mt = 0; mt < 2; ++mt) {
#pragma unroll
    for (int nt = 0; nt < 4; ++nt) {
      const int n = nloc + nt * 16 + l15;
#pragma unroll
      for (int r = 0; r < 8; ++r) {
        const int m = mOff + mt * 16 + hi * 8 + r;
        __builtin_nontemporal_store(acc[mt][nt][r],
                                    &base[(size_t)m * ldo + n]);
      }
    }
  }
}

// ---------------------------------------------------------------------------
// Launch
// ---------------------------------------------------------------------------
extern "C" void kernel_launch(void* const* d_in, const int* in_sizes, int n_in,
                              void* d_out, int out_size, void* d_ws, size_t ws_size,
                              hipStream_t stream) {
  const float* x     = (const float*)d_in[0];  // [8192, 4096]
  const float* gamma = (const float*)d_in[1];  // [4096]
  const float* wq    = (const float*)d_in[2];  // [4096, 4096]
  const float* wk    = (const float*)d_in[3];  // [4096, 1024]
  const float* wv    = (const float*)d_in[4];  // [4096, 1024]
  float* out = (float*)d_out;                  // q | k | v concatenated

  __bf16* xn = (__bf16*)d_ws;                  // 64 MB
  __bf16* wt = xn + (size_t)M_ROWS * K_DIM;    // 48 MB

  rmsnorm_bf16_kernel<<<M_ROWS, 256, 0, stream>>>(x, gamma, xn);
  wprep_kernel<<<dim3(N_TOT / 32, K_DIM / 32), 256, 0, stream>>>(wq, wk, wv, wt);
  qkv_gemm_kernel<<<dim3(N_TOT / 128, M_ROWS / 128), 256, 0, stream>>>(xn, wt, out);
}
